// MultiHeadSelfAttention_65231963291795
// MI455X (gfx1250) — matrix operand order repaired and checked by reading, not compiled, not run on hardware
//
#include <hip/hip_runtime.h>
#include <math.h>
#include <stdint.h>

#define BATCH 4
#define SEQ   2048
#define DIM   1024
#define NH    16
#define HD    64
#define MROWS (BATCH*SEQ)   // 8192

typedef __bf16 bf16_t;
typedef __attribute__((ext_vector_type(16))) __bf16        v16bf;
typedef __attribute__((ext_vector_type(8)))  float         v8f;
typedef __attribute__((ext_vector_type(4)))  unsigned int  u32x4;

union Frag16 { v16bf v; u32x4 u[2]; };

__device__ __forceinline__ bf16_t to_bf16(float f) {
    union { float f; unsigned int u; } in; in.f = f;
    unsigned int u = in.u;
    u += 0x7FFFu + ((u >> 16) & 1u);   // round-to-nearest-even
    union { unsigned short s; bf16_t b; } out;
    out.s = (unsigned short)(u >> 16);
    return out.b;
}

// ---------------------------------------------------------------- conversions
__global__ __launch_bounds__(256)
void cvt_f32_to_bf16(const float* __restrict__ src, bf16_t* __restrict__ dst, int n4) {
    int i = blockIdx.x * blockDim.x + threadIdx.x;
    if (i >= n4) return;
    float4 f = ((const float4*)src)[i];
    union { bf16_t h[4]; unsigned long long ll; } p;
    p.h[0] = to_bf16(f.x); p.h[1] = to_bf16(f.y);
    p.h[2] = to_bf16(f.z); p.h[3] = to_bf16(f.w);
    ((unsigned long long*)dst)[i] = p.ll;
}

// W[n][k] f32 -> Wt[k][n] bf16  (DIM x DIM)
__global__ __launch_bounds__(256)
void cvt_w_transpose(const float* __restrict__ W, bf16_t* __restrict__ Wt) {
    int idx = blockIdx.x * blockDim.x + threadIdx.x;   // n*DIM + k
    int n = idx >> 10;
    int k = idx & (DIM - 1);
    Wt[(size_t)k * DIM + n] = to_bf16(W[idx]);
}

// ---------------------------------------------------------------- GEMM (+RoPE)
// Y[m,n] = sum_k A[m,k] * W[n,k];  A:[MROWS,DIM] bf16, Wt:[DIM,DIM] bf16 (k-major)
// Each wave computes a 32x64 output macro-tile (2 M-tiles x 4 N-tiles).
// MODE 0: plain f32 out [MROWS,DIM]
// MODE 1: RoPE, bf16 out [B,H,S,HD]
// MODE 2: no RoPE, bf16 out [B,H,S,HD]
template<int MODE>
__global__ __launch_bounds__(256)
void proj_gemm(const bf16_t* __restrict__ A,
               const bf16_t* __restrict__ Wt,
               bf16_t* __restrict__ outH,
               float*  __restrict__ outF,
               const int* __restrict__ pos) {
    const int lane = threadIdx.x & 31;
    const int wave = threadIdx.x >> 5;
    const int tile = blockIdx.x * 8 + wave;      // 4096 macro-tiles total
    const int tm2 = tile >> 4;                   // 0..255 (32-row blocks)
    const int tn4 = tile & 15;                   // 0..15  (64-col blocks)

    const int half = lane >> 4;
    const int nl   = lane & 15;
    const int kg   = half ? 8 : 0;

    const bf16_t* arow0 = A + (size_t)(tm2 * 32 + nl)      * DIM;
    const bf16_t* arow1 = A + (size_t)(tm2 * 32 + 16 + nl) * DIM;

    v8f acc[2][4];
    #pragma unroll
    for (int mi = 0; mi < 2; ++mi)
        #pragma unroll
        for (int ni = 0; ni < 4; ++ni)
            acc[mi][ni] = (v8f){0.f,0.f,0.f,0.f,0.f,0.f,0.f,0.f};

    #pragma unroll 2
    for (int k0 = 0; k0 < DIM; k0 += 32) {
        Frag16 a0, a1;
        a0.u[0] = *(const u32x4*)(arow0 + k0 + kg);
        a0.u[1] = *(const u32x4*)(arow0 + k0 + 16 + kg);
        a1.u[0] = *(const u32x4*)(arow1 + k0 + kg);
        a1.u[1] = *(const u32x4*)(arow1 + k0 + 16 + kg);
        const bf16_t* brow = Wt + (size_t)k0 * DIM + tn4 * 64 + nl;
        if (k0 + 32 < DIM)
            __builtin_prefetch(brow + 32 * DIM, 0, 0);
        #pragma unroll
        for (int ni = 0; ni < 4; ++ni) {
            Frag16 bf;
            #pragma unroll
            for (int s = 0; s < 16; ++s)
                bf.v[s] = brow[(size_t)(s < 8 ? kg + s : 16 + kg + s - 8) * DIM + ni * 16];
            acc[0][ni] = __builtin_amdgcn_wmma_f32_16x16x32_bf16(false, a0.v, false, bf.v,
                                                                 (short)0, acc[0][ni], false, false);
            acc[1][ni] = __builtin_amdgcn_wmma_f32_16x16x32_bf16(false, a1.v, false, bf.v,
                                                                 (short)0, acc[1][ni], false, false);
        }
    }

    #pragma unroll
    for (int mi = 0; mi < 2; ++mi) {
        #pragma unroll
        for (int ni = 0; ni < 4; ++ni) {
            const int ncol = tn4 * 64 + ni * 16 + nl;
            if (MODE == 0) {
                #pragma unroll
                for (int r = 0; r < 8; ++r) {
                    int m = tm2 * 32 + mi * 16 + r + 8 * half;
                    outF[(size_t)m * DIM + ncol] = acc[mi][ni][r];
                }
            } else {
                const int h = ncol >> 6;
                const int d = ncol & 63;
                float inv = 0.f;
                if (MODE == 1) {
                    float p2 = (float)((d >> 1) * 2);
                    inv = __expf(-9.210340371976184f * p2 * (1.0f / 64.0f)); // theta^(-2p/hd)
                }
                #pragma unroll
                for (int r = 0; r < 8; ++r) {
                    int m = tm2 * 32 + mi * 16 + r + 8 * half;
                    int b = m >> 11;
                    int s = m & (SEQ - 1);
                    float val = acc[mi][ni][r];
                    if (MODE == 1) {
                        float ang = (float)pos[m] * inv;
                        float sn, cs;
                        __sincosf(ang, &sn, &cs);
                        float other = __shfl_xor(val, 1, 32);
                        val = ((d & 1) == 0) ? (val * cs - other * sn)
                                             : (other * sn + val * cs);
                    }
                    outH[(((size_t)b * NH + h) * SEQ + s) * HD + d] = to_bf16(val);
                }
            }
        }
    }
}

// ---------------------------------------------------------------- flash attention
// Q,K,V: bf16 [B,H,S,HD]; O: bf16 [B,S,DIM]. Causal, scale 1/8.
__global__ __launch_bounds__(256)
void flash_attn(const bf16_t* __restrict__ Q, const bf16_t* __restrict__ K,
                const bf16_t* __restrict__ V, bf16_t* __restrict__ O) {
    __shared__ __align__(16) bf16_t Kt[HD][32];      // [dim][key]  transposed
    __shared__ __align__(16) bf16_t Vt[32][HD];      // [key][dim]
    __shared__ __align__(16) bf16_t Pl[8][16][32];   // per-wave P tile

    const int lane = threadIdx.x & 31;
    const int wave = threadIdx.x >> 5;
    const int qt = blockIdx.x & 15;        // S/128
    const int bh = blockIdx.x >> 4;
    const int b  = bh >> 4;
    const int h  = bh & 15;
    const int q0 = qt * 128;

    const int half = lane >> 4;
    const int nl   = lane & 15;
    const int kg   = half ? 8 : 0;

    const size_t headBase = ((size_t)b * NH + h) * SEQ * HD;

    // Q fragments for this wave's 16 rows (reused across all key tiles)
    const bf16_t* qp = Q + headBase + (size_t)(q0 + wave * 16 + nl) * HD;
    Frag16 qa[2];
    #pragma unroll
    for (int c = 0; c < 2; ++c) {
        qa[c].u[0] = *(const u32x4*)(qp + c * 32 + kg);
        qa[c].u[1] = *(const u32x4*)(qp + c * 32 + 16 + kg);
    }

    v8f o[4];
    #pragma unroll
    for (int dc = 0; dc < 4; ++dc) o[dc] = (v8f){0.f,0.f,0.f,0.f,0.f,0.f,0.f,0.f};
    float mrow[8], lrow[8];
    #pragma unroll
    for (int r = 0; r < 8; ++r) { mrow[r] = -3.0e38f; lrow[r] = 0.f; }

    const int qmax_w = q0 + wave * 16 + 15;
    const int nkb = q0 / 32 + 4;

    // Per-thread staging coordinates: 32 keys x 64 dims per tile, 16B per thread
    const int skey = threadIdx.x >> 3;
    const int sd0  = (threadIdx.x & 7) * 8;
    const unsigned vlds = (unsigned)(uintptr_t)&Vt[skey][sd0];   // LDS byte address

    for (int jb = 0; jb < nkb; ++jb) {
        const int j0 = jb * 32;
        {
            // V tile: async DMA copy global -> LDS (tracked on ASYNCcnt)
            unsigned long long vga =
                (unsigned long long)(uintptr_t)(V + headBase + (size_t)(j0 + skey) * HD + sd0);
            asm volatile("global_load_async_to_lds_b128 %0, %1, off"
                         :: "v"(vlds), "v"(vga) : "memory");
            // K tile: load + transpose-store into LDS
            union { u32x4 u; bf16_t hh[8]; } ku;
            ku.u = *(const u32x4*)(K + headBase + (size_t)(j0 + skey) * HD + sd0);
            #pragma unroll
            for (int i = 0; i < 8; ++i) Kt[sd0 + i][skey] = ku.hh[i];
            asm volatile("s_wait_asynccnt 0x0" ::: "memory");
        }
        __syncthreads();

        if (j0 <= qmax_w) {              // wave-uniform: EXEC stays all-ones
            v8f s0 = {0.f,0.f,0.f,0.f,0.f,0.f,0.f,0.f};
            v8f s1 = {0.f,0.f,0.f,0.f,0.f,0.f,0.f,0.f};
            #pragma unroll
            for (int kc = 0; kc < 2; ++kc) {   // dim chunks 0-31 / 32-63
                Frag16 b0, b1;
                #pragma unroll
                for (int s = 0; s < 16; ++s) {
                    const int kk = kc * 32 + (s < 8 ? kg + s : 16 + kg + s - 8);
                    b0.v[s] = Kt[kk][nl];
                    b1.v[s] = Kt[kk][16 + nl];
                }
                s0 = __builtin_amdgcn_wmma_f32_16x16x32_bf16(false, qa[kc].v, false, b0.v,
                                                             (short)0, s0, false, false);
                s1 = __builtin_amdgcn_wmma_f32_16x16x32_bf16(false, qa[kc].v, false, b1.v,
                                                             (short)0, s1, false, false);
            }
            #pragma unroll
            for (int r = 0; r < 8; ++r) {
                const int qm = q0 + wave * 16 + r + 8 * half;
                float v0 = s0[r] * 0.125f, v1 = s1[r] * 0.125f;
                if (j0 + nl      > qm) v0 = -3.0e38f;
                if (j0 + 16 + nl > qm) v1 = -3.0e38f;
                float mj = fmaxf(v0, v1);
                mj = fmaxf(mj, __shfl_xor(mj, 1, 32));
                mj = fmaxf(mj, __shfl_xor(mj, 2, 32));
                mj = fmaxf(mj, __shfl_xor(mj, 4, 32));
                mj = fmaxf(mj, __shfl_xor(mj, 8, 32));
                float mnew = fmaxf(mrow[r], mj);
                float corr = __expf(mrow[r] - mnew);
                float p0 = __expf(v0 - mnew);
                float p1 = __expf(v1 - mnew);
                float ps = p0 + p1;
                ps += __shfl_xor(ps, 1, 32);
                ps += __shfl_xor(ps, 2, 32);
                ps += __shfl_xor(ps, 4, 32);
                ps += __shfl_xor(ps, 8, 32);
                lrow[r] = lrow[r] * corr + ps;
                mrow[r] = mnew;
                #pragma unroll
                for (int dc = 0; dc < 4; ++dc) o[dc][r] *= corr;
                const int ml = r + 8 * half;
                Pl[wave][ml][nl]      = to_bf16(p0);
                Pl[wave][ml][16 + nl] = to_bf16(p1);
            }
            // O += P * V   (P reloaded from LDS in A-fragment layout)
            Frag16 pa;
            pa.u[0] = *(const u32x4*)&Pl[wave][nl][kg];
            pa.u[1] = *(const u32x4*)&Pl[wave][nl][16 + kg];
            #pragma unroll
            for (int dc = 0; dc < 4; ++dc) {
                Frag16 bv;
                #pragma unroll
                for (int s = 0; s < 16; ++s)
                    bv.v[s] = Vt[s < 8 ? kg + s : 16 + kg + s - 8][dc * 16 + nl];
                o[dc] = __builtin_amdgcn_wmma_f32_16x16x32_bf16(false, pa.v, false, bv.v,
                                                                (short)0, o[dc], false, false);
            }
        }
        __syncthreads();
    }

    #pragma unroll
    for (int r = 0; r < 8; ++r) {
        const int srow = q0 + wave * 16 + r + 8 * half;
        const float inv = 1.0f / lrow[r];
        bf16_t* op = O + ((size_t)b * SEQ + srow) * DIM + h * HD;
        #pragma unroll
        for (int dc = 0; dc < 4; ++dc)
            op[dc * 16 + nl] = to_bf16(o[dc][r] * inv);
    }
}

// ---------------------------------------------------------------- launch
extern "C" void kernel_launch(void* const* d_in, const int* in_sizes, int n_in,
                              void* d_out, int out_size, void* d_ws, size_t ws_size,
                              hipStream_t stream) {
    const float* x   = (const float*)d_in[0];
    const int*   pos = (const int*)  d_in[1];
    const float* Wq  = (const float*)d_in[2];
    const float* Wk  = (const float*)d_in[3];
    const float* Wv  = (const float*)d_in[4];
    const float* Wo  = (const float*)d_in[5];
    float* out = (float*)d_out;

    char* ws = (char*)d_ws;
    size_t off = 0;
    auto alloc = [&](size_t bytes) -> void* {
        void* p = ws + off;
        off = (off + bytes + 255) & ~(size_t)255;
        return p;
    };
    bf16_t* Xbf  = (bf16_t*)alloc((size_t)MROWS * DIM * 2);
    bf16_t* Wqt  = (bf16_t*)alloc((size_t)DIM * DIM * 2);
    bf16_t* Wkt  = (bf16_t*)alloc((size_t)DIM * DIM * 2);
    bf16_t* Wvt  = (bf16_t*)alloc((size_t)DIM * DIM * 2);
    bf16_t* Wot  = (bf16_t*)alloc((size_t)DIM * DIM * 2);
    bf16_t* Qh   = (bf16_t*)alloc((size_t)MROWS * DIM * 2);
    bf16_t* Kh   = (bf16_t*)alloc((size_t)MROWS * DIM * 2);
    bf16_t* Vh   = (bf16_t*)alloc((size_t)MROWS * DIM * 2);
    bf16_t* Attn = (bf16_t*)alloc((size_t)MROWS * DIM * 2);

    cvt_f32_to_bf16<<<(MROWS * DIM / 4 + 255) / 256, 256, 0, stream>>>(x, Xbf, MROWS * DIM / 4);
    cvt_w_transpose<<<DIM * DIM / 256, 256, 0, stream>>>(Wq, Wqt);
    cvt_w_transpose<<<DIM * DIM / 256, 256, 0, stream>>>(Wk, Wkt);
    cvt_w_transpose<<<DIM * DIM / 256, 256, 0, stream>>>(Wv, Wvt);
    cvt_w_transpose<<<DIM * DIM / 256, 256, 0, stream>>>(Wo, Wot);

    const int gemmBlocks = (MROWS / 32) * (DIM / 64) / 8;   // 512
    proj_gemm<1><<<gemmBlocks, 256, 0, stream>>>(Xbf, Wqt, Qh, nullptr, pos);
    proj_gemm<1><<<gemmBlocks, 256, 0, stream>>>(Xbf, Wkt, Kh, nullptr, pos);
    proj_gemm<2><<<gemmBlocks, 256, 0, stream>>>(Xbf, Wvt, Vh, nullptr, pos);

    flash_attn<<<BATCH * NH * (SEQ / 128), 256, 0, stream>>>(Qh, Kh, Vh, Attn);

    proj_gemm<0><<<gemmBlocks, 256, 0, stream>>>(Attn, Wot, nullptr, out, nullptr);
}
